// ProbSparseSelfAttention_20186346291960
// MI455X (gfx1250) — compile-verified, matrix operand order checked by reading, not run on hardware
//
#include <hip/hip_runtime.h>
#include <hip/hip_bf16.h>

// ---------------------------------------------------------------------------
// ProbSparse self-attention for MI455X (gfx1250), wave32 + WMMA + async LDS.
// B=2, N=2048, C=1024, H=16, D=64, U=40.
//
// Pipeline:
//   1. cast query -> f16 ; cast+TRANSPOSE w_qkv, w_fc -> f16 [Nout][K]
//   2. WMMA GEMM (async double-buffered LDS): qkv = query @ w_qkv + b_qkv,
//      scattered into [3,B,H,N,D] f32
//   3. q row norms -> per-(b,h) top-40 -> scores softmax -> attn [B,H,U,N]
//   4. ctx = NaN-fill, then selected rows: second softmax(attn_row) @ v
//   5. WMMA GEMM: out = ctx_f16 @ w_fc + b_fc
// ---------------------------------------------------------------------------

typedef __attribute__((ext_vector_type(16))) _Float16 v16h;
typedef __attribute__((ext_vector_type(8)))  _Float16 v8h;
typedef __attribute__((ext_vector_type(8)))  float    v8f;

#define BN_SEQ 2048
#define HDIM   64
#define NHEAD  16
#define UTOP   40

// ---------------------------------------------------------------------------
// Async memory -> LDS copy, 16B per lane (GLOBAL_LOAD_ASYNC_TO_LDS_B128,
// tracked by ASYNCcnt).  Generic shared addresses carry the LDS byte offset
// in the low 32 bits (aperture in the high bits), so truncation yields the
// per-lane LDS address the instruction wants.
// ---------------------------------------------------------------------------
__device__ inline void async_copy_b128(const _Float16* gptr, _Float16* lptr) {
  unsigned lds = (unsigned)(uintptr_t)lptr;
  asm volatile("global_load_async_to_lds_b128 %0, %1, off"
               :: "v"(lds), "v"(gptr)
               : "memory");
}
__device__ inline void wait_async_le4() {
  asm volatile("s_wait_asynccnt 0x4" ::: "memory");
}
__device__ inline void wait_async_0() {
  asm volatile("s_wait_asynccnt 0x0" ::: "memory");
}

// ---------------------------------------------------------------------------
// Load a 16x32 f16 WMMA A/B fragment from LDS (row-major, stride in halves).
// ISA 7.12.2 16-bit layout: lane L -> row = L&15, kbase = (L>>4)*8;
// element e -> k = kbase+e (e<8) else kbase+e+8.  Two 16B-aligned ds_load_b128.
// ---------------------------------------------------------------------------
__device__ inline v16h load_frag(const _Float16* base, int row, int stride_h, int kbase) {
  const v8h lo = *(const v8h*)(base + (size_t)row * stride_h + kbase);
  const v8h hi = *(const v8h*)(base + (size_t)row * stride_h + kbase + 16);
  v16h r;
#pragma unroll
  for (int i = 0; i < 8; ++i) { r[i] = lo[i]; r[i + 8] = hi[i]; }
  return r;
}

// ---------------------------------------------------------------------------
// f16 WMMA GEMM: out = A[M,K] @ BT[Nout,K]^T + bias.
//   block = 256 threads (8 waves), tile 128(M) x 128(N), K-step 32,
//   double-buffered async LDS staging.
//   wave (wm,wn) owns 32x64: 2 A frags x 4 B frags -> 8 WMMAs / K-step.
// mode 0: out[row*Ndim + col]            (final FC)
// mode 1: scatter into qkv [3,B,H,N,D]   (QKV projection)
// ---------------------------------------------------------------------------
__global__ __launch_bounds__(256) void wmma_gemm_kernel(
    const _Float16* __restrict__ A, const _Float16* __restrict__ BT,
    const float* __restrict__ bias, float* __restrict__ out,
    int Kdim, int Ndim, int mode)
{
  __shared__ _Float16 As[2][128][40];  // 32 used + 8 pad per row
  __shared__ _Float16 Bs[2][128][40];  // rows of BT (n-major, k contiguous)

  const int t    = threadIdx.x;
  const int lane = t & 31;
  const int w    = t >> 5;
  const int wm   = w & 3;        // 0..3 -> M offset wm*32
  const int wn   = w >> 2;       // 0..1 -> N offset wn*64
  const int rowBase = blockIdx.y * 128;
  const int colBase = blockIdx.x * 128;
  const int kb8 = (lane >> 4) * 8;
  const int l16 = lane & 15;

  // per-thread staging chunks: 512 chunks of 8 halves for each of A and B
  const int ch0 = t * 2;
  const int r0  = ch0 >> 2,        c0 = (ch0 & 3) * 8;
  const int r1  = (ch0 + 1) >> 2,  c1 = ((ch0 + 1) & 3) * 8;

  v8f acc[2][4] = {};

  const int nk = Kdim >> 5;  // K-steps of 32

  // prologue: stage 0 -> buffer 0 (4 async ops per wave)
  {
    const int k0 = 0;
    async_copy_b128(A  + (size_t)(rowBase + r0) * Kdim + k0 + c0, &As[0][r0][c0]);
    async_copy_b128(BT + (size_t)(colBase + r0) * Kdim + k0 + c0, &Bs[0][r0][c0]);
    async_copy_b128(A  + (size_t)(rowBase + r1) * Kdim + k0 + c1, &As[0][r1][c1]);
    async_copy_b128(BT + (size_t)(colBase + r1) * Kdim + k0 + c1, &Bs[0][r1][c1]);
  }

  for (int s = 0; s < nk; ++s) {
    const int buf = s & 1;
    if (s + 1 < nk) {
      const int k0 = (s + 1) << 5;
      const int nb = buf ^ 1;
      async_copy_b128(A  + (size_t)(rowBase + r0) * Kdim + k0 + c0, &As[nb][r0][c0]);
      async_copy_b128(BT + (size_t)(colBase + r0) * Kdim + k0 + c0, &Bs[nb][r0][c0]);
      async_copy_b128(A  + (size_t)(rowBase + r1) * Kdim + k0 + c1, &As[nb][r1][c1]);
      async_copy_b128(BT + (size_t)(colBase + r1) * Kdim + k0 + c1, &Bs[nb][r1][c1]);
      wait_async_le4();   // stage s landed; stage s+1 still in flight
    } else {
      wait_async_0();
    }
    __syncthreads();

    v16h af[2], bf[4];
#pragma unroll
    for (int i = 0; i < 2; ++i)
      af[i] = load_frag(&As[buf][0][0], wm * 32 + i * 16 + l16, 40, kb8);
#pragma unroll
    for (int j = 0; j < 4; ++j)
      bf[j] = load_frag(&Bs[buf][0][0], wn * 64 + j * 16 + l16, 40, kb8);
#pragma unroll
    for (int i = 0; i < 2; ++i)
#pragma unroll
      for (int j = 0; j < 4; ++j)
        acc[i][j] = __builtin_amdgcn_wmma_f32_16x16x32_f16(
            false, af[i], false, bf[j], (short)0, acc[i][j], false, false);
    __syncthreads();  // safe to overwrite buf at stage s+2
  }

  // ---- epilogue: C/D layout: VGPR r -> M = r + (lane>>4)*8, N = lane&15 ----
#pragma unroll
  for (int tm = 0; tm < 2; ++tm) {
#pragma unroll
    for (int tn = 0; tn < 4; ++tn) {
      v8f cv   = acc[tm][tn];
      int gcol = colBase + wn * 64 + tn * 16 + l16;
      float bv = bias[gcol];
      int mb   = rowBase + wm * 32 + tm * 16 + (lane >> 4) * 8;
#pragma unroll
      for (int r = 0; r < 8; ++r) {
        int grow  = mb + r;
        float val = cv[r] + bv;
        if (mode == 0) {
          out[(size_t)grow * Ndim + gcol] = val;
        } else {
          int tt = gcol >> 10;
          int hh = (gcol >> 6) & (NHEAD - 1);
          int dd = gcol & (HDIM - 1);
          int bb = grow >> 11;
          int nn = grow & (BN_SEQ - 1);
          out[(((size_t)tt * 2 + bb) * NHEAD + hh) * BN_SEQ * HDIM +
              (size_t)nn * HDIM + dd] = val;
        }
      }
    }
  }
}

// ---------------------------------------------------------------------------
__global__ void qnorm_kernel(const float* __restrict__ q, float* __restrict__ norms) {
  int i = blockIdx.x * blockDim.x + threadIdx.x;  // 0..65535
  const float* qp = q + (size_t)i * HDIM;
  float s = 0.f;
#pragma unroll 8
  for (int d = 0; d < HDIM; ++d) { float x = qp[d]; s += x * x; }
  norms[i] = sqrtf(s);
}

// ---------------------------------------------------------------------------
__global__ __launch_bounds__(256) void topk_kernel(const float* __restrict__ norms,
                                                   int* __restrict__ topk) {
  __shared__ float vals[BN_SEQ];
  __shared__ float sval[256];
  __shared__ int   sidx[256];
  int bh = blockIdx.x, t = threadIdx.x;
  const float* src = norms + (size_t)bh * BN_SEQ;
  for (int i = t; i < BN_SEQ; i += 256) vals[i] = src[i];
  __syncthreads();
  for (int u = 0; u < UTOP; ++u) {
    float bv = -1e30f; int bi = 0;
    for (int i = t; i < BN_SEQ; i += 256)
      if (vals[i] > bv) { bv = vals[i]; bi = i; }
    sval[t] = bv; sidx[t] = bi;
    __syncthreads();
    for (int s = 128; s > 0; s >>= 1) {
      if (t < s && sval[t + s] > sval[t]) { sval[t] = sval[t + s]; sidx[t] = sidx[t + s]; }
      __syncthreads();
    }
    if (t == 0) { topk[bh * UTOP + u] = sidx[0]; vals[sidx[0]] = -1e30f; }
    __syncthreads();
  }
}

// ---------------------------------------------------------------------------
__global__ __launch_bounds__(256) void scores_softmax_kernel(
    const float* __restrict__ q, const float* __restrict__ k,
    const int* __restrict__ topk, float* __restrict__ attn)
{
  __shared__ float qrow[HDIM];
  __shared__ float sc[BN_SEQ];
  __shared__ float red[256];
  int blk = blockIdx.x, bh = blk / UTOP, u = blk % UTOP;
  int t = threadIdx.x;
  int row = topk[bh * UTOP + u];
  const float* qp = q + ((size_t)bh * BN_SEQ + row) * HDIM;
  if (t < HDIM) qrow[t] = qp[t];
  __syncthreads();
  const float* kb = k + (size_t)bh * BN_SEQ * HDIM;
  float lmax = -1e30f;
  for (int n = t; n < BN_SEQ; n += 256) {
    const float* kp = kb + (size_t)n * HDIM;
    float acc = 0.f;
#pragma unroll 8
    for (int d = 0; d < HDIM; ++d) acc += qrow[d] * kp[d];
    acc *= 0.125f;  // D^-0.5
    sc[n] = acc;
    lmax = fmaxf(lmax, acc);
  }
  red[t] = lmax; __syncthreads();
  for (int s = 128; s > 0; s >>= 1) { if (t < s) red[t] = fmaxf(red[t], red[t + s]); __syncthreads(); }
  float mx = red[0]; __syncthreads();
  float lsum = 0.f;
  for (int n = t; n < BN_SEQ; n += 256) lsum += __expf(sc[n] - mx);
  red[t] = lsum; __syncthreads();
  for (int s = 128; s > 0; s >>= 1) { if (t < s) red[t] += red[t + s]; __syncthreads(); }
  float inv = 1.f / red[0];
  float* ap = attn + ((size_t)bh * UTOP + u) * BN_SEQ;
  for (int n = t; n < BN_SEQ; n += 256) ap[n] = __expf(sc[n] - mx) * inv;
}

// ---------------------------------------------------------------------------
__global__ void fill_nan_kernel(float* __restrict__ p, int n) {
  int i = blockIdx.x * blockDim.x + threadIdx.x;
  union { unsigned u; float f; } nv; nv.u = 0x7FC00000u;
  if (i < n) p[i] = nv.f;
}

// ---------------------------------------------------------------------------
__global__ __launch_bounds__(256) void attnv_kernel(
    const float* __restrict__ attn, const float* __restrict__ v,
    const int* __restrict__ topk, float* __restrict__ ctx)
{
  __shared__ float pr[BN_SEQ];
  __shared__ float red[256];
  __shared__ float acc[4][HDIM];
  int blk = blockIdx.x, bh = blk / UTOP, u = blk % UTOP;
  int b = bh >> 4, h = bh & (NHEAD - 1);
  int t = threadIdx.x;
  int row = topk[bh * UTOP + u];
  const float* ap = attn + ((size_t)bh * UTOP + u) * BN_SEQ;
  for (int n = t; n < BN_SEQ; n += 256) pr[n] = ap[n];
  __syncthreads();
  float lmax = -1e30f;
  for (int n = t; n < BN_SEQ; n += 256) lmax = fmaxf(lmax, pr[n]);
  red[t] = lmax; __syncthreads();
  for (int s = 128; s > 0; s >>= 1) { if (t < s) red[t] = fmaxf(red[t], red[t + s]); __syncthreads(); }
  float mx = red[0]; __syncthreads();
  float lsum = 0.f;
  for (int n = t; n < BN_SEQ; n += 256) lsum += __expf(pr[n] - mx);
  red[t] = lsum; __syncthreads();
  for (int s = 128; s > 0; s >>= 1) { if (t < s) red[t] += red[t + s]; __syncthreads(); }
  float inv = 1.f / red[0];
  int d = t & (HDIM - 1), g = t >> 6;
  const float* vb = v + (size_t)bh * BN_SEQ * HDIM;
  float a0 = 0.f;
  for (int n = g; n < BN_SEQ; n += 4)
    a0 += __expf(pr[n] - mx) * inv * vb[(size_t)n * HDIM + d];
  acc[g][d] = a0; __syncthreads();
  if (g == 0) {
    float s0 = acc[0][d] + acc[1][d] + acc[2][d] + acc[3][d];
    ctx[((size_t)b * BN_SEQ + row) * 1024 + h * HDIM + d] = s0;
  }
}

// ---------------------------------------------------------------------------
__global__ void cast_f32_f16_kernel(const float* __restrict__ in,
                                    _Float16* __restrict__ out, int n) {
  int i = blockIdx.x * blockDim.x + threadIdx.x;
  if (i < n) out[i] = (_Float16)in[i];
}

// ---------------------------------------------------------------------------
// in: [K][N] f32  ->  out: [N][K] f16 (32x32 LDS tiles, coalesced both ways)
// ---------------------------------------------------------------------------
__global__ __launch_bounds__(256) void cast_transpose_kernel(
    const float* __restrict__ in, _Float16* __restrict__ out, int K, int N) {
  __shared__ float tile[32][33];
  int bx = blockIdx.x * 32;  // n
  int by = blockIdx.y * 32;  // k
  int tx = threadIdx.x & 31, ty = threadIdx.x >> 5;  // ty: 0..7
  for (int i = ty; i < 32; i += 8)
    tile[i][tx] = in[(size_t)(by + i) * N + bx + tx];
  __syncthreads();
  for (int i = ty; i < 32; i += 8)
    out[(size_t)(bx + i) * K + by + tx] = (_Float16)tile[tx][i];
}

// ---------------------------------------------------------------------------
extern "C" void kernel_launch(void* const* d_in, const int* in_sizes, int n_in,
                              void* d_out, int out_size, void* d_ws, size_t ws_size,
                              hipStream_t stream) {
  const float* query = (const float*)d_in[0];  // [B,N,C]
  const float* w_qkv = (const float*)d_in[1];  // [C,3C]
  const float* b_qkv = (const float*)d_in[2];  // [3C]
  const float* w_fc  = (const float*)d_in[3];  // [C,C]
  const float* b_fc  = (const float*)d_in[4];  // [C]
  float* out = (float*)d_out;                  // [B,N,C]

  char* ws = (char*)d_ws;
  size_t off = 0;
  auto alloc = [&](size_t bytes) -> void* {
    void* p = ws + off;
    off += (bytes + 255) & ~(size_t)255;
    return p;
  };

  const size_t BHND = (size_t)2 * NHEAD * BN_SEQ * HDIM;  // 4,194,304
  float*    qkv     = (float*)alloc(3 * BHND * sizeof(float));  // [3,B,H,N,D]
  float*    qmat    = qkv;
  float*    kmat    = qkv + BHND;
  float*    vmat    = qkv + 2 * BHND;
  _Float16* qf16    = (_Float16*)alloc((size_t)4194304 * 2);
  _Float16* wqkvT16 = (_Float16*)alloc((size_t)3145728 * 2);    // [3072][1024]
  _Float16* wfcT16  = (_Float16*)alloc((size_t)1048576 * 2);    // [1024][1024]
  float*    norms   = (float*)alloc((size_t)65536 * 4);
  int*      topk    = (int*)alloc((size_t)2 * NHEAD * UTOP * 4);
  float*    attn    = (float*)alloc((size_t)2 * NHEAD * UTOP * BN_SEQ * 4);
  float*    ctx     = (float*)alloc((size_t)4194304 * 4);
  _Float16* ctx16   = (_Float16*)alloc((size_t)4194304 * 2);

  // 1. operand casts (weights transposed to [Nout][K])
  cast_f32_f16_kernel<<<4194304 / 256, 256, 0, stream>>>(query, qf16, 4194304);
  cast_transpose_kernel<<<dim3(3072 / 32, 1024 / 32), 256, 0, stream>>>(w_qkv, wqkvT16, 1024, 3072);
  cast_transpose_kernel<<<dim3(1024 / 32, 1024 / 32), 256, 0, stream>>>(w_fc, wfcT16, 1024, 1024);

  // 2. QKV projection (WMMA, async double-buffered), scatter to [3,B,H,N,D]
  wmma_gemm_kernel<<<dim3(3072 / 128, 4096 / 128), 256, 0, stream>>>(
      qf16, wqkvT16, b_qkv, qkv, 1024, 3072, /*mode=*/1);

  // 3. ProbSparse attention
  qnorm_kernel<<<65536 / 256, 256, 0, stream>>>(qmat, norms);
  topk_kernel<<<2 * NHEAD, 256, 0, stream>>>(norms, topk);
  scores_softmax_kernel<<<2 * NHEAD * UTOP, 256, 0, stream>>>(qmat, kmat, topk, attn);
  fill_nan_kernel<<<4194304 / 256, 256, 0, stream>>>(ctx, 4194304);
  attnv_kernel<<<2 * NHEAD * UTOP, 256, 0, stream>>>(attn, vmat, topk, ctx);

  // 4. final FC (WMMA)
  cast_f32_f16_kernel<<<4194304 / 256, 256, 0, stream>>>(ctx, ctx16, 4194304);
  wmma_gemm_kernel<<<dim3(1024 / 128, 4096 / 128), 256, 0, stream>>>(
      ctx16, wfcT16, b_fc, out, 1024, 1024, /*mode=*/0);
}
